// Mymodel_66030827209097
// MI455X (gfx1250) — matrix operand order repaired and checked by reading, not compiled, not run on hardware
//
#include <hip/hip_runtime.h>
#include <math.h>

// ---------------------------------------------------------------------------
// DGCNN-like network for MI455X (gfx1250, wave32).
//  * Gram + top-k fused (f32 VALU, register insertion lists, LDS tiling with
//    gfx1250 GLOBAL_LOAD_ASYNC_TO_LDS_B32 staging + s_wait_asynccnt)
//  * 1x1 conv + BN + leaky via v_wmma_f32_16x16x32_f16 (f16 in, f32 acc),
//    branch-free operand loads (K % 32 == 0), 32x32 wave macro-tile:
//    4 WMMAs / K-step from 4 fragments (A0,A1,B0,B1)
//  * concat-free layout: stages write into channel slices of xcat
// ---------------------------------------------------------------------------

#define NEGS     0.2f
#define BN_SCALE 0.9999950000374996f   // 1/sqrt(1+1e-5)
#define LN_EPS   1e-5f

#define NB   16      // batch
#define NP   2048    // points
#define KNN  20      // k (reference passes k=20; fixed at compile time)

typedef __attribute__((ext_vector_type(16))) _Float16 v16h;
typedef __attribute__((ext_vector_type(8)))  float    v8f;

__device__ __forceinline__ float lrelu(float v) { return v > 0.f ? v : NEGS * v; }

// gfx1250 async global->LDS copy (ASYNCcnt-tracked); per-lane scatter.
__device__ __forceinline__ void async_load_f32_to_lds(const float* gptr, float* lptr) {
#if defined(__gfx1250__)
  asm volatile("global_load_async_to_lds_b32 %0, %1, off"
               :
               : "v"((unsigned)(unsigned long long)(uintptr_t)lptr),
                 "v"((unsigned long long)(uintptr_t)gptr)
               : "memory");
#else
  *lptr = *gptr;
#endif
}
__device__ __forceinline__ void wait_async() {
#if defined(__gfx1250__)
  asm volatile("s_wait_asynccnt 0x0" ::: "memory");
#endif
}

// --------------------------------------------------------------------------
// xx[n] = sum_c x[c,n]^2
// --------------------------------------------------------------------------
__global__ void xx_kernel(const float* __restrict__ in, long inBS, int C,
                          float* __restrict__ xx) {
  int n = blockIdx.x * blockDim.x + threadIdx.x;
  int b = blockIdx.y;
  const float* p = in + (long)b * inBS + n;
  float s = 0.f;
  for (int c = 0; c < C; ++c) { float v = p[(long)c * NP]; s = fmaf(v, v, s); }
  xx[b * NP + n] = s;
}

// --------------------------------------------------------------------------
// Fused pairwise-distance + top-k. One thread per row n; 64-column LDS tiles
// staged with async global->LDS. pd[n,m] = 2<xn,xm> - |xn|^2 - |xm|^2 ;
// keep the 20 largest (skip m==n). Insertion list fully unrolled -> VGPRs.
// --------------------------------------------------------------------------
template <int C>
__global__ __launch_bounds__(256) void topk_kernel(const float* __restrict__ in, long inBS,
                                                   const float* __restrict__ xx,
                                                   int* __restrict__ idxout) {
  constexpr int TM = 64;
  __shared__ float xt[TM * C];             // column tile, [mm][c]
  int tid = threadIdx.x;
  int n = blockIdx.x * 256 + tid;
  int b = blockIdx.y;
  const float* xb = in + (long)b * inBS;

  float q[C];
#pragma unroll
  for (int c = 0; c < C; ++c) q[c] = xb[(long)c * NP + n];
  float xxn = xx[b * NP + n];

  float val[KNN]; int ind[KNN];
#pragma unroll
  for (int j = 0; j < KNN; ++j) { val[j] = -3.0e38f; ind[j] = 0; }

  for (int m0 = 0; m0 < NP; m0 += TM) {
    __syncthreads();
    for (int t = tid; t < C * TM; t += 256) {
      int mm = t % TM, c = t / TM;                     // coalesced global reads
      async_load_f32_to_lds(xb + (long)c * NP + m0 + mm, &xt[mm * C + c]);
    }
    {
      // prefetch next tile into cache (gfx1250 global_prefetch_b8)
      int pm = tid % TM, pc = tid / TM;
      if (m0 + TM < NP && pc < C)
        __builtin_prefetch(xb + (long)pc * NP + m0 + TM + pm, 0, 1);
    }
    wait_async();
    __syncthreads();
    for (int mm = 0; mm < TM; ++mm) {
      int m = m0 + mm;
      float dot = 0.f;
#pragma unroll
      for (int c = 0; c < C; ++c) dot = fmaf(q[c], xt[mm * C + c], dot);  // LDS broadcast
      float pd = 2.f * dot - xxn - xx[b * NP + m];
      if (m != n && pd > val[KNN - 1]) {
        float cv = pd; int ci = m;
#pragma unroll
        for (int j = 0; j < KNN; ++j) {
          if (cv > val[j]) {
            float tv = val[j]; int ti = ind[j];
            val[j] = cv; ind[j] = ci; cv = tv; ci = ti;
          }
        }
      }
    }
  }
  int* o = idxout + ((long)b * NP + n) * KNN;
#pragma unroll
  for (int j = 0; j < KNN; ++j) o[j] = ind[j];
}

// --------------------------------------------------------------------------
// Gather-k attention: s_j = <q, key_j>, softmax over j, feat = sum w_j*(key_j-q).
// Output h = [q ; feat] as (B, 2C, N). Gathers hit L2 (x slice <= 1 MB/batch).
// --------------------------------------------------------------------------
template <int C>
__global__ __launch_bounds__(256) void attn_kernel(const float* __restrict__ in, long inBS,
                                                   const int* __restrict__ idx,
                                                   float* __restrict__ h) {
  int n = blockIdx.x * 256 + threadIdx.x;
  int b = blockIdx.y;
  const float* xb = in + (long)b * inBS;
  const int* ip = idx + ((long)b * NP + n) * KNN;
  int nb[KNN];
#pragma unroll
  for (int j = 0; j < KNN; ++j) nb[j] = ip[j];

  float s[KNN];
#pragma unroll
  for (int j = 0; j < KNN; ++j) s[j] = 0.f;
  for (int c = 0; c < C; ++c) {
    const float* row = xb + (long)c * NP;
    float qc = row[n];
#pragma unroll
    for (int j = 0; j < KNN; ++j) s[j] = fmaf(qc, row[nb[j]], s[j]);
  }
  float mx = s[0];
#pragma unroll
  for (int j = 1; j < KNN; ++j) mx = fmaxf(mx, s[j]);
  float sum = 0.f;
#pragma unroll
  for (int j = 0; j < KNN; ++j) { s[j] = __expf(s[j] - mx); sum += s[j]; }
  float inv = 1.f / sum;
#pragma unroll
  for (int j = 0; j < KNN; ++j) s[j] *= inv;

  float* hb = h + (long)b * (2 * C) * NP;
  for (int c = 0; c < C; ++c) {
    const float* row = xb + (long)c * NP;
    float qc = row[n];
    float f = 0.f;
#pragma unroll
    for (int j = 0; j < KNN; ++j) f = fmaf(s[j], row[nb[j]] - qc, f);
    hb[(long)c * NP + n]       = qc;   // concat([q, feat]) -> channels [0,C) = q
    hb[(long)(C + c) * NP + n] = f;    // channels [C,2C) = feat
  }
}

// --------------------------------------------------------------------------
// Layer-1 conv (K=6, O=64): tiny, scalar, weights staged in LDS.
// --------------------------------------------------------------------------
__global__ __launch_bounds__(256) void conv1_kernel(
    const float* __restrict__ in,   // h stage 1: (B, 6, N)
    const float* __restrict__ W,    // (64, 6)
    const float* __restrict__ gam, const float* __restrict__ bet,
    float* __restrict__ out, long outBS) {
  __shared__ float Ws[64 * 6];
  __shared__ float gs[64], bs[64];
  int t = threadIdx.x;
  if (t < 64 * 6) Ws[t] = W[t];
  if (t < 64) { gs[t] = gam[t] * BN_SCALE; bs[t] = bet[t]; }
  __syncthreads();
  int n = blockIdx.x * 256 + t;
  int b = blockIdx.y;
  const float* hb = in + (long)b * 6 * NP + n;
  float x0 = hb[0],        x1 = hb[NP],     x2 = hb[2 * NP];
  float x3 = hb[3 * NP],   x4 = hb[4 * NP], x5 = hb[5 * NP];
  float* ob = out + (long)b * outBS + n;
  for (int o = 0; o < 64; ++o) {
    const float* w = Ws + o * 6;
    float y = w[0]*x0 + w[1]*x1 + w[2]*x2 + w[3]*x3 + w[4]*x4 + w[5]*x5;
    ob[(long)o * NP] = lrelu(y * gs[o] + bs[o]);
  }
}

// --------------------------------------------------------------------------
// 1x1 conv + BN affine + leaky via WMMA f32<-f16 16x16x32.
// Precondition: K % 32 == 0, O % 32 == 0  ->  ALL loads are branch-free.
// One wave computes a 32(o) x 32(n) macro-tile: 4 WMMAs per K-step from
// 4 fragments (A0,A1 share B0,B1) -> 16 scalar loads per WMMA.
// Operand packing per CDNA5 ISA 7.12.2:
//   A (16x32 f16): lane m=lane&15, half=lane>>4; VGPR v: K = (v/4)*16 + half*8 + 2*(v&3)
//   B (32x16 f16): lane n=lane&15, khalf=lane>>4; VGPR v: K = (v/4)*16 + khalf*8 + 2*(v&3)
//   C/D (16x16 f32): M = vgpr + 8*(lane>>4), N = lane&15
// --------------------------------------------------------------------------
__global__ __launch_bounds__(128) void conv_bn_wmma_kernel(
    const float* __restrict__ in, long inBS, int K,
    const float* __restrict__ W, const float* __restrict__ gam,
    const float* __restrict__ bet, float* __restrict__ out, long outBS) {
  int wave = threadIdx.x >> 5;
  int lane = threadIdx.x & 31;
  int n0 = blockIdx.x * 128 + wave * 32;   // wave covers 32 points
  int o0 = blockIdx.y * 32;                // wave covers 32 output channels
  int b  = blockIdx.z;
  int m    = lane & 15;
  int hsel = lane >> 4;

  const float* Wrow0 = W + (long)(o0 + m) * K;        // A tile 0 row
  const float* Wrow1 = Wrow0 + (long)16 * K;          // A tile 1 row
  const float* col0  = in + (long)b * inBS + (n0 + m);// B tile 0 column
  const float* col1  = col0 + 16;                     // B tile 1 column

  v8f a00 = {}, a01 = {}, a10 = {}, a11 = {};
  for (int k0 = 0; k0 < K; k0 += 32) {
    v16h A0, A1, B0, B1;
#pragma unroll
    for (int v = 0; v < 8; ++v) {
      int grp = v >> 2, vv = v & 3;
      int ka = k0 + grp * 16 + hsel * 8 + 2 * vv;
      A0[2 * v]     = (_Float16)Wrow0[ka];
      A0[2 * v + 1] = (_Float16)Wrow0[ka + 1];
      A1[2 * v]     = (_Float16)Wrow1[ka];
      A1[2 * v + 1] = (_Float16)Wrow1[ka + 1];
      int kb = k0 + grp * 16 + hsel * 8 + 2 * vv;
      B0[2 * v]     = (_Float16)col0[(long)kb * NP];
      B0[2 * v + 1] = (_Float16)col0[(long)(kb + 1) * NP];
      B1[2 * v]     = (_Float16)col1[(long)kb * NP];
      B1[2 * v + 1] = (_Float16)col1[(long)(kb + 1) * NP];
    }
    a00 = __builtin_amdgcn_wmma_f32_16x16x32_f16(false, A0, false, B0, (short)0, a00, false, false);
    a01 = __builtin_amdgcn_wmma_f32_16x16x32_f16(false, A0, false, B1, (short)0, a01, false, false);
    a10 = __builtin_amdgcn_wmma_f32_16x16x32_f16(false, A1, false, B0, (short)0, a10, false, false);
    a11 = __builtin_amdgcn_wmma_f32_16x16x32_f16(false, A1, false, B1, (short)0, a11, false, false);
  }
#pragma unroll
  for (int v = 0; v < 8; ++v) {
    int o = o0 + v + 8 * hsel;
    float sc0 = BN_SCALE * gam[o],      bb0 = bet[o];
    float sc1 = BN_SCALE * gam[o + 16], bb1 = bet[o + 16];
    float* op0 = out + (long)b * outBS + (long)o * NP + (n0 + m);
    float* op1 = op0 + (long)16 * NP;
    op0[0]  = lrelu(a00[v] * sc0 + bb0);
    op0[16] = lrelu(a01[v] * sc0 + bb0);
    op1[0]  = lrelu(a10[v] * sc1 + bb1);
    op1[16] = lrelu(a11[v] * sc1 + bb1);
  }
}

// --------------------------------------------------------------------------
// scores[b,n,h] = lrelu(sum_e x5[b,e,n] * Watt[h,e]),  H=4, E=1024
// --------------------------------------------------------------------------
__global__ void scores_kernel(const float* __restrict__ x5,
                              const float* __restrict__ Watt,
                              float* __restrict__ sc) {
  int n = blockIdx.x * blockDim.x + threadIdx.x;
  int b = blockIdx.y;
  const float* xb = x5 + (long)b * 1024 * NP;
  float a0 = 0, a1 = 0, a2 = 0, a3 = 0;
  for (int e = 0; e < 1024; ++e) {
    float v = xb[(long)e * NP + n];             // coalesced across threads
    a0 = fmaf(v, Watt[e],        a0);
    a1 = fmaf(v, Watt[1024 + e], a1);
    a2 = fmaf(v, Watt[2048 + e], a2);
    a3 = fmaf(v, Watt[3072 + e], a3);
  }
  float* o = sc + ((long)b * NP + n) * 4;
  o[0] = lrelu(a0); o[1] = lrelu(a1); o[2] = lrelu(a2); o[3] = lrelu(a3);
}

// --------------------------------------------------------------------------
// att[b, h*1024+e] = sum_n x5[b,e,n] * scores[b,n,h]
// --------------------------------------------------------------------------
__global__ void att_kernel(const float* __restrict__ x5,
                           const float* __restrict__ sc,
                           float* __restrict__ att) {
  int e = blockIdx.x * blockDim.x + threadIdx.x;
  int b = blockIdx.y;
  const float* row = x5 + ((long)b * 1024 + e) * NP;   // per-thread unit stride
  const float* sb  = sc + (long)b * NP * 4;
  float a0 = 0, a1 = 0, a2 = 0, a3 = 0;
  for (int n = 0; n < NP; ++n) {
    float v = row[n];
    const float* s = sb + n * 4;
    a0 = fmaf(v, s[0], a0); a1 = fmaf(v, s[1], a1);
    a2 = fmaf(v, s[2], a2); a3 = fmaf(v, s[3], a3);
  }
  float* o = att + (long)b * 4096;
  o[e] = a0; o[1024 + e] = a1; o[2048 + e] = a2; o[3072 + e] = a3;
}

// --------------------------------------------------------------------------
// LayerNorm(4096) + leaky, then 3-layer MLP head. One workgroup per batch.
// --------------------------------------------------------------------------
__global__ __launch_bounds__(512) void head_kernel(
    const float* __restrict__ att,
    const float* __restrict__ ln_g, const float* __restrict__ ln_b,
    const float* __restrict__ Wl1, const float* __restrict__ bl1,
    const float* __restrict__ g6,  const float* __restrict__ b6,
    const float* __restrict__ Wl2, const float* __restrict__ bl2,
    const float* __restrict__ g7,  const float* __restrict__ b7,
    const float* __restrict__ Wl3, const float* __restrict__ bl3,
    float* __restrict__ out) {
  __shared__ float a[4096];
  __shared__ float y1[512];
  __shared__ float y2[256];
  __shared__ float red[512];
  int t = threadIdx.x, b = blockIdx.x;
  const float* ab = att + (long)b * 4096;

  float s = 0.f, s2 = 0.f;
  for (int i = t; i < 4096; i += 512) {
    float v = ab[i]; a[i] = v; s += v; s2 = fmaf(v, v, s2);
  }
  red[t] = s; __syncthreads();
  for (int off = 256; off > 0; off >>= 1) { if (t < off) red[t] += red[t + off]; __syncthreads(); }
  float mean = red[0] * (1.f / 4096.f); __syncthreads();
  red[t] = s2; __syncthreads();
  for (int off = 256; off > 0; off >>= 1) { if (t < off) red[t] += red[t + off]; __syncthreads(); }
  float var  = red[0] * (1.f / 4096.f) - mean * mean;
  float rstd = rsqrtf(var + LN_EPS);
  __syncthreads();
  for (int i = t; i < 4096; i += 512)
    a[i] = lrelu((a[i] - mean) * rstd * ln_g[i] + ln_b[i]);
  __syncthreads();
  {
    const float* w = Wl1 + (long)t * 4096;
    float acc = 0.f;
    for (int i = 0; i < 4096; ++i) acc = fmaf(w[i], a[i], acc);
    y1[t] = lrelu((acc + bl1[t]) * BN_SCALE * g6[t] + b6[t]);
  }
  __syncthreads();
  if (t < 256) {
    const float* w = Wl2 + (long)t * 512;
    float acc = 0.f;
    for (int i = 0; i < 512; ++i) acc = fmaf(w[i], y1[i], acc);
    y2[t] = lrelu((acc + bl2[t]) * BN_SCALE * g7[t] + b7[t]);
  }
  __syncthreads();
  if (t < 40) {
    const float* w = Wl3 + (long)t * 256;
    float acc = 0.f;
    for (int i = 0; i < 256; ++i) acc = fmaf(w[i], y2[i], acc);
    out[b * 40 + t] = acc + bl3[t];
  }
}

// ---------------------------------------------------------------------------
extern "C" void kernel_launch(void* const* d_in, const int* in_sizes, int n_in,
                              void* d_out, int out_size, void* d_ws, size_t ws_size,
                              hipStream_t stream) {
  (void)in_sizes; (void)n_in; (void)out_size; (void)ws_size;
  const float* x    = (const float*)d_in[0];
  const float* W1   = (const float*)d_in[1];
  const float* g1   = (const float*)d_in[2];
  const float* b1   = (const float*)d_in[3];
  const float* W2   = (const float*)d_in[4];
  const float* g2   = (const float*)d_in[5];
  const float* b2   = (const float*)d_in[6];
  const float* W3   = (const float*)d_in[7];
  const float* g3   = (const float*)d_in[8];
  const float* b3   = (const float*)d_in[9];
  const float* W4   = (const float*)d_in[10];
  const float* g4   = (const float*)d_in[11];
  const float* b4   = (const float*)d_in[12];
  const float* W5   = (const float*)d_in[13];
  const float* g5   = (const float*)d_in[14];
  const float* b5   = (const float*)d_in[15];
  const float* Watt = (const float*)d_in[16];
  const float* ln_g = (const float*)d_in[17];
  const float* ln_b = (const float*)d_in[18];
  const float* Wl1  = (const float*)d_in[19];
  const float* bl1  = (const float*)d_in[20];
  const float* g6   = (const float*)d_in[21];
  const float* b6   = (const float*)d_in[22];
  const float* Wl2  = (const float*)d_in[23];
  const float* bl2  = (const float*)d_in[24];
  const float* g7   = (const float*)d_in[25];
  const float* b7   = (const float*)d_in[26];
  const float* Wl3  = (const float*)d_in[27];
  const float* bl3  = (const float*)d_in[28];
  // d_in[29] = k (==20, baked in as KNN)

  // ---- workspace carve-up (float-aligned regions) ----
  char* ws = (char*)d_ws;
  float* xx   = (float*)ws; ws += sizeof(float) * (size_t)NB * NP;             // 128 KB
  int*   idx  = (int*)  ws; ws += sizeof(int)   * (size_t)NB * NP * KNN;       // 2.5 MB
  float* h    = (float*)ws; ws += sizeof(float) * (size_t)NB * 256 * NP;       // 32 MB (max 2C)
  float* xcat = (float*)ws; ws += sizeof(float) * (size_t)NB * 512 * NP;       // 64 MB
  float* x5   = (float*)ws; ws += sizeof(float) * (size_t)NB * 1024 * NP;      // 128 MB
  float* sc   = (float*)ws; ws += sizeof(float) * (size_t)NB * NP * 4;         // 512 KB
  float* attb = (float*)ws; ws += sizeof(float) * (size_t)NB * 4096;           // 256 KB

  dim3 blk256(256), grdN(NP / 256, NB);
  const long bsX = (long)3 * NP, bsC = (long)512 * NP;

  // ---- stage 1: C=3, conv 6->64 (scalar kernel) -> xcat rows [0,64)
  xx_kernel      <<<grdN, blk256, 0, stream>>>(x, bsX, 3, xx);
  topk_kernel<3> <<<grdN, blk256, 0, stream>>>(x, bsX, xx, idx);
  attn_kernel<3> <<<grdN, blk256, 0, stream>>>(x, bsX, idx, h);
  conv1_kernel   <<<grdN, blk256, 0, stream>>>(h, W1, g1, b1, xcat, bsC);

  // ---- stage 2: C=64 (xcat rows [0,64)), conv 128->64 -> xcat rows [64,128)
  const float* in2 = xcat;
  xx_kernel       <<<grdN, blk256, 0, stream>>>(in2, bsC, 64, xx);
  topk_kernel<64> <<<grdN, blk256, 0, stream>>>(in2, bsC, xx, idx);
  attn_kernel<64> <<<grdN, blk256, 0, stream>>>(in2, bsC, idx, h);
  conv_bn_wmma_kernel<<<dim3(NP / 128, 64 / 32, NB), 128, 0, stream>>>(
      h, (long)128 * NP, 128, W2, g2, b2, xcat + (long)64 * NP, bsC);

  // ---- stage 3: C=64 (xcat rows [64,128)), conv 128->128 -> xcat rows [128,256)
  const float* in3 = xcat + (long)64 * NP;
  xx_kernel       <<<grdN, blk256, 0, stream>>>(in3, bsC, 64, xx);
  topk_kernel<64> <<<grdN, blk256, 0, stream>>>(in3, bsC, xx, idx);
  attn_kernel<64> <<<grdN, blk256, 0, stream>>>(in3, bsC, idx, h);
  conv_bn_wmma_kernel<<<dim3(NP / 128, 128 / 32, NB), 128, 0, stream>>>(
      h, (long)128 * NP, 128, W3, g3, b3, xcat + (long)128 * NP, bsC);

  // ---- stage 4: C=128 (xcat rows [128,256)), conv 256->256 -> xcat rows [256,512)
  const float* in4 = xcat + (long)128 * NP;
  xx_kernel        <<<grdN, blk256, 0, stream>>>(in4, bsC, 128, xx);
  topk_kernel<128> <<<grdN, blk256, 0, stream>>>(in4, bsC, xx, idx);
  attn_kernel<128> <<<grdN, blk256, 0, stream>>>(in4, bsC, idx, h);
  conv_bn_wmma_kernel<<<dim3(NP / 128, 256 / 32, NB), 128, 0, stream>>>(
      h, (long)256 * NP, 256, W4, g4, b4, xcat + (long)256 * NP, bsC);

  // ---- conv5: 512 -> 1024 on xcat (concat is free by construction)
  conv_bn_wmma_kernel<<<dim3(NP / 128, 1024 / 32, NB), 128, 0, stream>>>(
      xcat, bsC, 512, W5, g5, b5, x5, (long)1024 * NP);

  // ---- global attention pooling + LN + MLP head
  scores_kernel<<<grdN, blk256, 0, stream>>>(x5, Watt, sc);
  att_kernel<<<dim3(1024 / 256, NB), 256, 0, stream>>>(x5, sc, attb);
  head_kernel<<<NB, 512, 0, stream>>>(attb, ln_g, ln_b, Wl1, bl1, g6, b6,
                                      Wl2, bl2, g7, b7, Wl3, bl3, (float*)d_out);
}
